// EfficientSelfAttention_89017492177455
// MI455X (gfx1250) — matrix operand order repaired and checked by reading, not compiled, not run on hardware
//
#include <hip/hip_runtime.h>
#include <cstddef>

// ---------------- types ----------------
typedef __bf16 bf16_t;
typedef __attribute__((ext_vector_type(16))) __bf16 v16bf;
typedef __attribute__((ext_vector_type(8)))  float  v8f;

struct alignas(16) U128 { unsigned int w[4]; };

union FragB {          // one WMMA 16x16x32 bf16 operand: 16 bf16 per lane
    v16bf v;
    U128  q[2];
};

static __device__ __forceinline__ v8f wmma_bf16(const FragB& a, const FragB& b, v8f c) {
    return __builtin_amdgcn_wmma_f32_16x16x32_bf16(
        /*neg_a=*/false, a.v, /*neg_b=*/false, b.v,
        /*c_mod=*/(short)0, c, /*reuse_a=*/false, /*reuse_b=*/false);
}

// ---------------- problem constants ----------------
#define BATCH 4
#define CDIM  512
#define HWDIM 4096      // 64*64
#define HEADS 8
#define DHEAD 64
#define LK    1024      // 32*32 keys
#define KPATCH 2048     // 512*2*2

// ---------------- elementwise fp32 -> bf16 ----------------
__global__ void cvt_bf16(const float* __restrict__ in, bf16_t* __restrict__ out, int n) {
    int i = blockIdx.x * blockDim.x + threadIdx.x;
    if (i < n) out[i] = (bf16_t)in[i];
}

// tiled transpose [rows][cols] fp32 -> [cols][rows] bf16, per batch (blockIdx.z)
__global__ void transpose_bf16(const float* __restrict__ in, bf16_t* __restrict__ out,
                               int rows, int cols) {
    __shared__ float tile[32][33];
    size_t b = blockIdx.z;
    in  += b * (size_t)rows * cols;
    out += b * (size_t)rows * cols;
    int c0 = blockIdx.x * 32, r0 = blockIdx.y * 32;
    #pragma unroll
    for (int i = threadIdx.y; i < 32; i += 8)
        tile[i][threadIdx.x] = in[(size_t)(r0 + i) * cols + c0 + threadIdx.x];
    __syncthreads();
    #pragma unroll
    for (int i = threadIdx.y; i < 32; i += 8)
        out[(size_t)(c0 + i) * rows + r0 + threadIdx.x] = (bf16_t)tile[threadIdx.x][i];
}

// unfold 2x2/stride-2 patches: PT[b][n=1024][k=2048], k = c*4 + i*2 + j, n = y*32+x
__global__ void build_patches(const float* __restrict__ x, bf16_t* __restrict__ PT) {
    size_t idx = (size_t)blockIdx.x * blockDim.x + threadIdx.x;
    if (idx >= (size_t)BATCH * LK * KPATCH) return;
    int k  = idx & (KPATCH - 1);
    int n  = (idx >> 11) & (LK - 1);
    int b  = idx >> 21;
    int c = k >> 2, i = (k >> 1) & 1, j = k & 1;
    int y = n >> 5, xx = n & 31;
    PT[idx] = (bf16_t)x[(((size_t)b * CDIM + c) * 64 + (2 * y + i)) * 64 + (2 * xx + j)];
}

// ---------------- WMMA GEMM: C[M,N] = A[M,K] * BT[N,K]^T ----------------
// MODE 0: store bf16 per-head transposed [B][8][N][64], pre-scaled by 1/8 (q projection;
//         0.125 is a power of two -> exact in bf16, removes scaling from the attention loop)
// MODE 1: rows <512 -> k like MODE0 (unscaled); rows >=512 -> vT [B][8][64][1024]
// MODE 2: store fp32 [B][M][N] directly                  (output projection)
template <int MODE>
__global__ __launch_bounds__(256)
void gemm_wmma(const bf16_t* __restrict__ A, const bf16_t* __restrict__ BT,
               void* __restrict__ out0, void* __restrict__ out1,
               int M, int N, int K) {
    const int lane = threadIdx.x & 31;
    const int w    = threadIdx.x >> 5;
    const int wm = w & 1, wn = w >> 1;                    // 2x4 wave grid -> 64x128 block tile
    const size_t b = blockIdx.z;
    const int m0 = blockIdx.y * 64 + wm * 32;
    const int n0 = blockIdx.x * 128 + wn * 32;
    BT += b * (size_t)N * K;

    const int lm = lane & 15;
    const int ah = (lane >> 4) * 8;    // A half: K offset {0,8} (+16 for second chunk)
    const int bh = (lane >> 4) * 8;    // B half: K offset {0,8} (+16 for second chunk)

    const bf16_t* aRow0 = A  + (size_t)(m0 + lm) * K;
    const bf16_t* aRow1 = aRow0 + (size_t)16 * K;
    const bf16_t* bRow0 = BT + (size_t)(n0 + lm) * K;
    const bf16_t* bRow1 = bRow0 + (size_t)16 * K;

    v8f acc[2][2] = {};
    #pragma unroll 2
    for (int kk = 0; kk < K; kk += 32) {
        FragB a0, a1, b0, b1;
        a0.q[0] = *(const U128*)(aRow0 + kk + ah);
        a0.q[1] = *(const U128*)(aRow0 + kk + ah + 16);
        a1.q[0] = *(const U128*)(aRow1 + kk + ah);
        a1.q[1] = *(const U128*)(aRow1 + kk + ah + 16);
        b0.q[0] = *(const U128*)(bRow0 + kk + bh);
        b0.q[1] = *(const U128*)(bRow0 + kk + bh + 16);
        b1.q[0] = *(const U128*)(bRow1 + kk + bh);
        b1.q[1] = *(const U128*)(bRow1 + kk + bh + 16);
        if (kk + 256 < K) {                       // global_prefetch_b8 ahead in K
            __builtin_prefetch(aRow0 + kk + 256, 0, 1);
            __builtin_prefetch(bRow0 + kk + 256, 0, 1);
        }
        acc[0][0] = wmma_bf16(a0, b0, acc[0][0]);
        acc[0][1] = wmma_bf16(a0, b1, acc[0][1]);
        acc[1][0] = wmma_bf16(a1, b0, acc[1][0]);
        acc[1][1] = wmma_bf16(a1, b1, acc[1][1]);
    }

    const int hi = (lane >> 4) * 8;   // C/D layout: VGPR r holds row r (lanes 0-15) / r+8 (16-31)
    #pragma unroll
    for (int tm = 0; tm < 2; ++tm) {
        const int ob = m0 + tm * 16;
        #pragma unroll
        for (int tn = 0; tn < 2; ++tn) {
            const int p = n0 + tn * 16 + lm;
            v8f& c = acc[tm][tn];
            if (MODE == 2) {
                float* O = (float*)out0 + b * (size_t)M * N;
                #pragma unroll
                for (int r = 0; r < 8; ++r)
                    O[(size_t)(ob + hi + r) * N + p] = c[r];
            } else {
                const int oh  = (MODE == 1) ? (ob & 511) : ob;
                const int head = oh >> 6;
                const int dd0  = (oh & 63) + hi;
                const bool isV = (MODE == 1) && (ob >= 512);
                if (!isV) {   // per-head [L=N][64], 8 consecutive dd -> one 16B store
                    const float s = (MODE == 0) ? 0.125f : 1.0f;   // fold d^-0.5 into q
                    bf16_t* O = (bf16_t*)out0;
                    union { bf16_t h[8]; U128 q; } pk;
                    #pragma unroll
                    for (int r = 0; r < 8; ++r) pk.h[r] = (bf16_t)(c[r] * s);
                    *(U128*)(O + (((b * HEADS + head) * (size_t)N + p) * DHEAD + dd0)) = pk.q;
                } else {      // vT [B][8][64][1024]
                    bf16_t* O = (bf16_t*)out1;
                    #pragma unroll
                    for (int r = 0; r < 8; ++r)
                        O[((b * HEADS + head) * (size_t)DHEAD + dd0 + r) * LK + p] = (bf16_t)c[r];
                }
            }
        }
    }
}

// ---------------- fused flash attention ----------------
// q(pre-scaled) [B][8][4096][64], k [B][8][1024][64], vT [B][8][64][1024]
//   -> oT [B][4096][512] (bf16)
__global__ __launch_bounds__(128)
void attn_flash(const bf16_t* __restrict__ qb, const bf16_t* __restrict__ kb,
                const bf16_t* __restrict__ vtb, bf16_t* __restrict__ oT) {
    __shared__ bf16_t pstage[4][16][32];   // per-wave prob-tile relayout buffer
    const int lane = threadIdx.x & 31;
    const int w    = threadIdx.x >> 5;
    const int h = blockIdx.y;
    const size_t b = blockIdx.z;
    const int pbase = (blockIdx.x * 4 + w) * 16;
    const int lm   = lane & 15;
    const int hi8  = (lane >> 4) * 8;
    const int hi16 = (lane >> 4) * 16;

    const bf16_t* qp = qb  + (b * HEADS + h) * (size_t)HWDIM * DHEAD;
    const bf16_t* kp = kb  + (b * HEADS + h) * (size_t)LK * DHEAD;
    const bf16_t* vp = vtb + (b * HEADS + h) * (size_t)DHEAD * LK;

    FragB qf[2];   // 16 query rows x full d=64, resident for the whole key loop
    {
        const bf16_t* qr = qp + (size_t)(pbase + lm) * DHEAD;
        qf[0].q[0] = *(const U128*)(qr + hi8);
        qf[0].q[1] = *(const U128*)(qr + hi8 + 16);
        qf[1].q[0] = *(const U128*)(qr + 32 + hi8);
        qf[1].q[1] = *(const U128*)(qr + 32 + hi8 + 16);
    }

    v8f acc[4] = {};
    float mrun[8], lrun[8];
    #pragma unroll
    for (int r = 0; r < 8; ++r) { mrun[r] = -3.0e38f; lrun[r] = 0.f; }

    #pragma unroll 2
    for (int j0 = 0; j0 < LK; j0 += 32) {
        // ---- issue ALL global loads for this j-block up front ----
        FragB kf[2][2];                  // [j-tile][d-chunk]
        #pragma unroll
        for (int jt = 0; jt < 2; ++jt)
            #pragma unroll
            for (int t = 0; t < 2; ++t) {
                const bf16_t* kr = kp + (size_t)(j0 + jt * 16 + lm) * DHEAD + t * 32 + hi8;
                kf[jt][t].q[0] = *(const U128*)(kr);
                kf[jt][t].q[1] = *(const U128*)(kr + 16);
            }
        FragB vf[4];                     // independent of softmax -> overlaps sim+softmax
        #pragma unroll
        for (int t = 0; t < 4; ++t) {
            const bf16_t* vr = vp + (size_t)(t * 16 + lm) * LK + j0 + hi8;
            vf[t].q[0] = *(const U128*)(vr);
            vf[t].q[1] = *(const U128*)(vr + 16);
        }

        // ---- sim = q @ k^T (q pre-scaled by 1/8); alternate accs to dodge RAW NOPs ----
        v8f s0 = {}, s1 = {};
        s0 = wmma_bf16(qf[0], kf[0][0], s0);
        s1 = wmma_bf16(qf[0], kf[1][0], s1);
        s0 = wmma_bf16(qf[1], kf[0][1], s0);
        s1 = wmma_bf16(qf[1], kf[1][1], s1);

        // ---- online softmax over the 16x32 score block (fully unrolled: 8
        //      independent cross-lane reduction chains interleave) ----
        float corr[8];
        #pragma unroll
        for (int r = 0; r < 8; ++r) {
            float a = s0[r], c2 = s1[r];
            float mx = fmaxf(a, c2);
            #pragma unroll
            for (int d = 1; d < 16; d <<= 1) mx = fmaxf(mx, __shfl_xor(mx, d, 16));
            float nm = fmaxf(mrun[r], mx);
            float p0 = __expf(a - nm), p1 = __expf(c2 - nm);
            float rs = p0 + p1;
            #pragma unroll
            for (int d = 1; d < 16; d <<= 1) rs += __shfl_xor(rs, d, 16);
            float cr = __expf(mrun[r] - nm);
            lrun[r] = lrun[r] * cr + rs;
            mrun[r] = nm;
            corr[r] = cr;
            const int row = r + hi8;               // C-layout row for this lane
            pstage[w][row][lm]      = (bf16_t)p0;
            pstage[w][row][16 + lm] = (bf16_t)p1;
        }
        #pragma unroll
        for (int t = 0; t < 4; ++t)
            #pragma unroll
            for (int r = 0; r < 8; ++r) acc[t][r] *= corr[r];

        // same-wave LDS is in-order; fence the compiler + wait DScnt
        asm volatile("s_wait_dscnt 0" ::: "memory");
        FragB pf;                                  // probs relayout: C-layout -> A-layout
        pf.q[0] = *(const U128*)&pstage[w][lm][hi8];
        pf.q[1] = *(const U128*)&pstage[w][lm][hi8 + 16];

        // ---- out += P @ V, 4 independent accumulators ----
        acc[0] = wmma_bf16(pf, vf[0], acc[0]);
        acc[1] = wmma_bf16(pf, vf[1], acc[1]);
        acc[2] = wmma_bf16(pf, vf[2], acc[2]);
        acc[3] = wmma_bf16(pf, vf[3], acc[3]);
    }

    // normalize and write oT[b][p][h*64 + dd]  (== transposed-B operand of out-proj GEMM)
    bf16_t* op = oT + b * (size_t)HWDIM * CDIM;
    #pragma unroll
    for (int t = 0; t < 4; ++t) {
        const int c = h * DHEAD + t * 16 + lm;
        #pragma unroll
        for (int r = 0; r < 8; ++r) {
            float o = acc[t][r] / lrun[r];
            op[(size_t)(pbase + hi8 + r) * CDIM + c] = (bf16_t)o;
        }
    }
}

// ---------------- host orchestration ----------------
extern "C" void kernel_launch(void* const* d_in, const int* in_sizes, int n_in,
                              void* d_out, int out_size, void* d_ws, size_t ws_size,
                              hipStream_t stream) {
    const float* x    = (const float*)d_in[0];
    const float* wq   = (const float*)d_in[1];
    const float* wkv  = (const float*)d_in[2];
    const float* wout = (const float*)d_in[3];

    size_t off = 0;
    auto alloc = [&](size_t bytes) {
        char* p = (char*)d_ws + off;
        off += (bytes + 255) & ~(size_t)255;
        return p;
    };
    bf16_t* wqb   = (bf16_t*)alloc((size_t)CDIM * CDIM * 2);
    bf16_t* wkvb  = (bf16_t*)alloc((size_t)1024 * KPATCH * 2);
    bf16_t* woutb = (bf16_t*)alloc((size_t)CDIM * CDIM * 2);
    bf16_t* xT    = (bf16_t*)alloc((size_t)BATCH * HWDIM * CDIM * 2);
    bf16_t* PT    = (bf16_t*)alloc((size_t)BATCH * LK * KPATCH * 2);
    bf16_t* qbuf  = (bf16_t*)alloc((size_t)BATCH * HEADS * HWDIM * DHEAD * 2);
    bf16_t* kbuf  = (bf16_t*)alloc((size_t)BATCH * HEADS * LK * DHEAD * 2);
    bf16_t* vtbuf = (bf16_t*)alloc((size_t)BATCH * HEADS * DHEAD * LK * 2);
    bf16_t* oT    = (bf16_t*)alloc((size_t)BATCH * HWDIM * CDIM * 2);
    if (off > ws_size) return;   // ~81 MB needed

    // precision conversion / repacking
    cvt_bf16<<<dim3((CDIM * CDIM + 255) / 256), 256, 0, stream>>>(wq, wqb, CDIM * CDIM);
    cvt_bf16<<<dim3((1024 * KPATCH + 255) / 256), 256, 0, stream>>>(wkv, wkvb, 1024 * KPATCH);
    cvt_bf16<<<dim3((CDIM * CDIM + 255) / 256), 256, 0, stream>>>(wout, woutb, CDIM * CDIM);
    transpose_bf16<<<dim3(HWDIM / 32, CDIM / 32, BATCH), dim3(32, 8), 0, stream>>>(x, xT, CDIM, HWDIM);
    build_patches<<<dim3((unsigned)(((size_t)BATCH * LK * KPATCH + 255) / 256)), 256, 0, stream>>>(x, PT);

    // q projection: [512,4096] = wq[512,512] * xT^T   (output pre-scaled by 1/8)
    gemm_wmma<0><<<dim3(HWDIM / 128, CDIM / 64, BATCH), 256, 0, stream>>>(
        wqb, xT, qbuf, nullptr, CDIM, HWDIM, CDIM);
    // kv projection (2x2 s2 conv as GEMM): [1024,1024] = wkv[1024,2048] * PT^T
    gemm_wmma<1><<<dim3(LK / 128, 1024 / 64, BATCH), 256, 0, stream>>>(
        wkvb, PT, kbuf, vtbuf, 1024, LK, KPATCH);
    // fused softmax(q k^T) v   (scale already folded into q)
    attn_flash<<<dim3(HWDIM / 64, HEADS, BATCH), 128, 0, stream>>>(qbuf, kbuf, vtbuf, oT);
    // output projection straight into d_out (fp32)
    gemm_wmma<2><<<dim3(HWDIM / 128, CDIM / 64, BATCH), 256, 0, stream>>>(
        woutb, oT, d_out, nullptr, CDIM, HWDIM, CDIM);
}
